// get_model_15779709846289
// MI455X (gfx1250) — matrix operand order repaired and checked by reading, not compiled, not run on hardware
//
#include <hip/hip_runtime.h>
#include <hip/hip_bf16.h>

typedef _Float16 half_t;
typedef __attribute__((ext_vector_type(16))) _Float16 v16h;
typedef __attribute__((ext_vector_type(8)))  _Float16 v8h;
typedef __attribute__((ext_vector_type(8)))  float    v8f;

#define BATCH 8
#define NPTS  8192
#define NCLS  50
#define CDIV(a,b) (((a)+(b)-1)/(b))

static inline int rup_h(int x, int m){ return ((x + m - 1) / m) * m; }

// ---------------------------------------------------------------- WMMA utils
__device__ inline v8f zero8(){
  v8f v;
#pragma unroll
  for (int i=0;i<8;i++) v[i]=0.f;
  return v;
}
__device__ inline v16h combine16(v8h lo, v8h hi){
  v16h r;
#pragma unroll
  for (int i=0;i<8;i++){ r[i]=lo[i]; r[8+i]=hi[i]; }
  return r;
}
// A fragment from LDS row (row = lane&15). halves [k0+kg*8 .. +8) and [k0+16+kg*8 .. +8)
__device__ inline v16h ldA(const half_t* rowk0, int kg){
  const v8h* p = (const v8h*)(rowk0 + kg*8);
  return combine16(p[0], p[2]);
}
// B fragment: Wt row (col = lane&15). halves [k0+kg*8 .. +8) and [k0+16+kg*8 .. +8)
__device__ inline v16h ldB(const half_t* rowk0, int kg){
  const v8h* p = (const v8h*)(rowk0 + kg*8);
  return combine16(p[0], p[2]);
}
__device__ inline v8f wmma32(v16h a, v16h b, v8f c){
  return __builtin_amdgcn_wmma_f32_16x16x32_f16(false, a, false, b, (short)0, c, false, false);
}

// ---------------------------------------------------------------- small kernels
__global__ void k_split(const float* __restrict__ in, float* __restrict__ pxyz,
                        float* __restrict__ pcol, int N){
  int i = blockIdx.x*blockDim.x + threadIdx.x;
  if (i >= BATCH*N) return;
  int b = i / N, n = i - b*N;
#pragma unroll
  for (int c=0;c<3;c++){
    pxyz[(size_t)i*3+c] = in[((size_t)b*6 + c)*N + n];
    pcol[(size_t)i*3+c] = in[((size_t)b*6 + 3 + c)*N + n];
  }
}

// farthest point sampling; one block per batch; dist array in LDS
__global__ void k_fps(const float* __restrict__ xyz, int* __restrict__ out, int N, int S){
  __shared__ float dist[8192];
  __shared__ float rv[256];
  __shared__ int   ri[256];
  __shared__ int   curS;
  int b = blockIdx.x, tid = threadIdx.x;
  for (int i=tid;i<N;i+=256) dist[i] = 1e10f;
  if (tid==0){ out[(size_t)b*S] = 0; curS = 0; }
  __syncthreads();
  for (int it=1; it<S; ++it){
    int cur = curS;
    const float* c = xyz + ((size_t)b*N + cur)*3;
    float cx=c[0], cy=c[1], cz=c[2];
    float bm=-1.f; int bi=0;
    for (int i=tid;i<N;i+=256){
      const float* p = xyz + ((size_t)b*N + i)*3;
      float dx=p[0]-cx, dy=p[1]-cy, dz=p[2]-cz;
      float d = dx*dx+dy*dy+dz*dz;
      float nd = fminf(dist[i], d);
      dist[i] = nd;
      if (nd > bm){ bm = nd; bi = i; }
    }
    rv[tid]=bm; ri[tid]=bi;
    __syncthreads();
    for (int s2=128;s2>0;s2>>=1){
      if (tid<s2){
        float ov=rv[tid+s2]; int oi=ri[tid+s2];
        if (ov>rv[tid] || (ov==rv[tid] && oi<ri[tid])){ rv[tid]=ov; ri[tid]=oi; }
      }
      __syncthreads();
    }
    if (tid==0){ out[(size_t)b*S+it]=ri[0]; curS=ri[0]; }
    __syncthreads();
  }
}

__global__ void k_gather3(const float* __restrict__ xyz, const int* __restrict__ fps,
                          float* __restrict__ ctr, int N, int S){
  int i = blockIdx.x*blockDim.x + threadIdx.x;
  if (i >= BATCH*S) return;
  int b = i / S;
  int j = fps[i];
  const float* p = xyz + ((size_t)b*N + j)*3;
#pragma unroll
  for (int c=0;c<3;c++) ctr[(size_t)i*3+c]=p[c];
}

// ball query: first K indices (index order) with d2<=r2, pad with first
__global__ void k_qball(const float* __restrict__ xyz, const float* __restrict__ ctr,
                        int* __restrict__ out, int N, int S, int K, float r2){
  int i = blockIdx.x*blockDim.x + threadIdx.x;
  if (i >= BATCH*S) return;
  int b = i / S;
  const float* c = ctr + (size_t)i*3;
  float cx=c[0], cy=c[1], cz=c[2];
  int* o = out + (size_t)i*K;
  int cnt=0;
  for (int n=0; n<N && cnt<K; n++){
    const float* p = xyz + ((size_t)b*N + n)*3;
    float dx=p[0]-cx, dy=p[1]-cy, dz=p[2]-cz;
    if (dx*dx+dy*dy+dz*dz <= r2) o[cnt++] = n;
  }
  int first = cnt>0 ? o[0] : 0;
  for (; cnt<K; cnt++) o[cnt]=first;
}

// weight convert: Wt[padOut][kpad] (f16, transposed, zero-padded) from W[Cin][Cout]
__global__ void k_wconv(const float* __restrict__ W, half_t* __restrict__ Wt,
                        int Cin, int Cout, int kpad, int padOut){
  int i = blockIdx.x*blockDim.x + threadIdx.x;
  if (i >= padOut*kpad) return;
  int o = i / kpad, k = i - o*kpad;
  float v = (o<Cout && k<Cin) ? W[(size_t)k*Cout + o] : 0.f;
  Wt[i] = (half_t)v;
}

__global__ void k_stconv(const float* __restrict__ b, const float* __restrict__ g,
                         const float* __restrict__ beta, float* __restrict__ s,
                         float* __restrict__ t, int Cout, int padOut, int mode){
  int i = blockIdx.x*blockDim.x + threadIdx.x;
  if (i >= padOut) return;
  const float inv = 0.99999500003749968755f; // 1/sqrt(1+1e-5)
  float sv=0.f, tv=0.f;
  if (i < Cout){
    if (mode==0){ sv = g[i]*inv; tv = b[i]*g[i]*inv + beta[i]; }
    else        { sv = 1.f;      tv = b[i]; }
  }
  s[i]=sv; t[i]=tv;
}

// 3-NN inverse-distance interpolation
__global__ void k_interp3(const float* __restrict__ x1, const float* __restrict__ x2,
                          const float* __restrict__ f2, float* __restrict__ out,
                          int N1, int N2, int C){
  int i = blockIdx.x*blockDim.x + threadIdx.x;
  if (i >= BATCH*N1) return;
  int b = i / N1;
  const float* p = x1 + (size_t)i*3;
  float px=p[0], py=p[1], pz=p[2];
  float d0=1e30f,d1=1e30f,d2=1e30f; int i0=0,i1=0,i2=0;
  for (int j=0;j<N2;j++){
    const float* q = x2 + ((size_t)b*N2 + j)*3;
    float dx=q[0]-px, dy=q[1]-py, dz=q[2]-pz;
    float d = dx*dx+dy*dy+dz*dz;
    if (d < d0){ d2=d1;i2=i1; d1=d0;i1=i0; d0=d;i0=j; }
    else if (d < d1){ d2=d1;i2=i1; d1=d;i1=j; }
    else if (d < d2){ d2=d;i2=j; }
  }
  float w0=1.f/(d0+1e-8f), w1=1.f/(d1+1e-8f), w2=1.f/(d2+1e-8f);
  float ws=w0+w1+w2; w0/=ws; w1/=ws; w2/=ws;
  const float* r0 = f2 + ((size_t)b*N2 + i0)*C;
  const float* r1 = f2 + ((size_t)b*N2 + i1)*C;
  const float* r2 = f2 + ((size_t)b*N2 + i2)*C;
  float* ro = out + (size_t)i*C;
  for (int c=0;c<C;c++) ro[c] = w0*r0[c] + w1*r1[c] + w2*r2[c];
}

__global__ void k_maxpool(const float* __restrict__ in, float* __restrict__ out, int K, int C){
  int i = blockIdx.x*blockDim.x + threadIdx.x;
  if (i >= BATCH*C) return;
  int b = i / C, c = i - b*C;
  float m = -1e30f;
  for (int r=0;r<K;r++){
    float v = in[((size_t)b*K + r)*C + c];
    m = v>m ? v : m;
  }
  out[i] = m;
}

__global__ void k_logsoftmax(const float* __restrict__ in, float* __restrict__ out, int M, int C){
  int i = blockIdx.x*blockDim.x + threadIdx.x;
  if (i >= M) return;
  const float* r = in + (size_t)i*C;
  float m = -1e30f;
  for (int c=0;c<C;c++) m = fmaxf(m, r[c]);
  float s = 0.f;
  for (int c=0;c<C;c++) s += expf(r[c]-m);
  float ls = logf(s);
  float* o = out + (size_t)i*C;
  for (int c=0;c<C;c++) o[c] = r[c] - m - ls;
}

__global__ void k_l3copy(const float* __restrict__ p, const float* __restrict__ c,
                         float* __restrict__ out){
  int i = blockIdx.x*blockDim.x + threadIdx.x;
  if (i >= BATCH*2048) return;
  int b = i / 2048, k = i - b*2048;
  out[i] = (k < 1024) ? p[(size_t)b*1024 + k] : c[(size_t)b*1024 + k - 1024];
}

// ---------------------------------------------------------------- fused SA branch
struct FLayer { const half_t* Wt; const float* s; const float* t; int Cin, Cout, kpad, outW; };
struct FArgs  { FLayer L[3]; int nL; };

// one workgroup per (center s, batch b): gather K neighbors (xyz-rel + src feats) as
// f16 into LDS, run up to 3 WMMA MLP layers LDS->LDS, max-pool over K, store (b,s,:)
__global__ void __launch_bounds__(128) k_sa_fused(
    const float* __restrict__ xyz, const float* __restrict__ src, int Csrc,
    const int* __restrict__ idx, const float* __restrict__ ctr,
    float* __restrict__ out, int outStride, int outOff,
    int N, int S, int K, int Wmax, FArgs A)
{
  extern __shared__ half_t sm[];
  half_t* buf0 = sm;
  half_t* buf1 = sm + (size_t)K*Wmax;
  int b = blockIdx.y, s = blockIdx.x;
  int tid = threadIdx.x, nth = blockDim.x;
  const int* myidx = idx + ((size_t)b*S + s)*K;
  const float* cc = ctr + ((size_t)b*S + s)*3;
  float cx=cc[0], cy=cc[1], cz=cc[2];

  int Cin0 = A.L[0].Cin;
  int W0   = A.L[0].kpad;
  for (int e=tid; e<K*W0; e+=nth){
    int r = e / W0, c = e - r*W0;
    float v = 0.f;
    if (c < Cin0){
      int j = myidx[r];
      if (j < 0) j = 0;
      if (j >= N) j = N-1;
      if (c < 3){
        float pv = xyz[((size_t)b*N + j)*3 + c];
        v = pv - (c==0 ? cx : (c==1 ? cy : cz));
      } else {
        v = src[((size_t)b*N + j)*Csrc + (c-3)];
      }
    }
    buf0[e] = (half_t)v;
  }
  __syncthreads();

  half_t* ib = buf0;
  half_t* ob = buf1;
  int wave = tid >> 5, lane = tid & 31, nwaves = nth >> 5;
  int kg = lane >> 4, ln = lane & 15;

  for (int l=0; l<A.nL; l++){
    FLayer Ly = A.L[l];
    int inW = Ly.kpad, outW = Ly.outW;
    int mtiles = K >> 4, ntiles = outW >> 4;
    for (int t0=wave; t0<mtiles*ntiles; t0+=nwaves){
      int mt = t0 / ntiles, nt = t0 - mt*ntiles;
      int row = mt*16 + ln, col = nt*16 + ln;
      v8f acc = zero8();
      const half_t* ar = ib + (size_t)row*inW;
      const half_t* br = Ly.Wt + (size_t)col*inW;
      for (int k0=0;k0<inW;k0+=32){
        v16h a  = ldA(ar + k0, kg);
        v16h bb = ldB(br + k0, kg);
        acc = wmma32(a, bb, acc);
      }
      float sc = (col < Ly.Cout) ? Ly.s[col] : 0.f;
      float tc = (col < Ly.Cout) ? Ly.t[col] : 0.f;
#pragma unroll
      for (int r=0;r<8;r++){
        float y = acc[r]*sc + tc;
        y = y > 0.f ? y : 0.f;
        ob[(size_t)(mt*16 + r + kg*8)*outW + col] = (half_t)y;
      }
    }
    __syncthreads();
    half_t* tmp = ib; ib = ob; ob = tmp;
  }

  int Cl = A.L[A.nL-1].Cout, Wl = A.L[A.nL-1].outW;
  for (int c=tid; c<Cl; c+=nth){
    float m = -1e30f;
    for (int r=0;r<K;r++){
      float v = (float)ib[(size_t)r*Wl + c];
      m = v>m ? v : m;
    }
    out[((size_t)b*S + s)*outStride + outOff + c] = m;
  }
}

// ---------------------------------------------------------------- generic WMMA GEMM
// Y[M][Cout] = epi( concat(Xa[M][Ca], Xb[.][Cb]) @ W ), weights pre-packed f16 Wt[padOut][kpad]
__device__ inline float fetch2(const float* ra, int Ca, const float* rb, int Cb, int k){
  if (k < Ca) return ra[k];
  int k2 = k - Ca;
  if (k2 < Cb) return rb[k2];
  return 0.f;
}

__global__ void __launch_bounds__(256) k_gemm(
    const float* __restrict__ Xa, int Ca, const float* __restrict__ Xb, int Cb, int xbDiv,
    const half_t* __restrict__ Wt, int kpad,
    const float* __restrict__ s, const float* __restrict__ t,
    float* __restrict__ Y, int M, int Cout, int doRelu)
{
  int wave = threadIdx.x >> 5, lane = threadIdx.x & 31;
  int gm = (blockIdx.x*8 + wave)*16;
  if (gm >= M) return;
  int kg = lane >> 4, ln = lane & 15;
  int nb = blockIdx.y*128;
  int ntiles = (Cout - nb + 15) >> 4; if (ntiles > 8) ntiles = 8;

  int row = gm + ln; if (row >= M) row = M-1;
  const float* ra = Xa + (size_t)row*Ca;
  const float* rb = Xb ? Xb + (size_t)(xbDiv ? (row/xbDiv) : row)*Cb : nullptr;

  v8f acc[8];
#pragma unroll
  for (int nt=0;nt<8;nt++) acc[nt]=zero8();

  for (int k0=0;k0<kpad;k0+=32){
    v16h a;
#pragma unroll
    for (int i=0;i<8;i++){
      int k1 = k0 + kg*8 + i;
      a[i]   = (half_t)fetch2(ra,Ca,rb,Cb,k1);
      a[8+i] = (half_t)fetch2(ra,Ca,rb,Cb,k1+16);
    }
#pragma unroll
    for (int nt=0;nt<8;nt++){
      if (nt < ntiles){
        int col = nb + nt*16 + ln;
        v16h bb = ldB(Wt + (size_t)col*kpad + k0, kg);
        acc[nt] = wmma32(a, bb, acc[nt]);
      }
    }
  }
#pragma unroll
  for (int nt=0;nt<8;nt++){
    if (nt < ntiles){
      int col = nb + nt*16 + ln;
      float sc = (col < Cout) ? s[col] : 0.f;
      float tc = (col < Cout) ? t[col] : 0.f;
#pragma unroll
      for (int r=0;r<8;r++){
        int rowo = gm + r + kg*8;
        if (rowo < M && col < Cout){
          float y = acc[nt][r]*sc + tc;
          if (doRelu) y = y > 0.f ? y : 0.f;
          Y[(size_t)rowo*Cout + col] = y;
        }
      }
    }
  }
}

// ---------------------------------------------------------------- host orchestration
struct LayerH { const float* W; const float* b; const float* g; const float* beta;
                int Cin,Cout,kpad,padOut,mode; size_t wOff,sOff; };

extern "C" void kernel_launch(void* const* d_in, const int* in_sizes, int n_in,
                              void* d_out, int out_size, void* d_ws, size_t ws_size,
                              hipStream_t stream)
{
  (void)in_sizes; (void)n_in; (void)out_size; (void)ws_size;
  auto F = [&](int i)->const float* { return (const float*)d_in[i]; };

  // ---- walk param leaves (jax dict tree order: sorted keys; per-layer b,beta,g,w)
  LayerH L[64]; int nLt = 0;
  size_t wCur = 0, sCur = 0;
  int leaf = 2;
  auto addBN = [&](int Cin, int Cout)->int{
    LayerH& x = L[nLt];
    x.b=F(leaf); x.beta=F(leaf+1); x.g=F(leaf+2); x.W=F(leaf+3); leaf+=4;
    x.Cin=Cin; x.Cout=Cout; x.kpad=rup_h(Cin,32); x.padOut=rup_h(Cout,32); x.mode=0;
    x.wOff=wCur; wCur += (size_t)x.kpad * x.padOut;
    x.sOff=sCur; sCur += (size_t)2 * x.padOut;
    return nLt++;
  };
  int conv2L;
  { // conv2: keys b, w
    LayerH& x = L[nLt];
    x.b=F(leaf); x.W=F(leaf+1); leaf+=2; x.g=x.b; x.beta=x.b;
    x.Cin=128; x.Cout=NCLS; x.kpad=128; x.padOut=rup_h(NCLS,32); x.mode=1;
    x.wOff=wCur; wCur += (size_t)x.kpad*x.padOut;
    x.sOff=sCur; sCur += (size_t)2*x.padOut;
    conv2L = nLt++;
  }
  int fp1_c[2] = { addBN(131,128), addBN(128,128) };
  int fp1_p[2] = { addBN(131,128), addBN(128,128) };
  int fp2_c[2] = { addBN(576,256), addBN(256,128) };
  int fp2_p[2] = { addBN(576,256), addBN(256,128) };
  int fp3_c[2] = { addBN(1536,256), addBN(256,256) };
  int fp3_p[2] = { addBN(1536,256), addBN(256,256) };
  int head0    = addBN(256,128);
  static const int SA1D[3][4] = {{6,32,32,64},{6,64,64,128},{6,64,96,128}};
  static const int SA2D[2][4] = {{323,128,128,256},{323,128,196,256}};
  static const int SA3D[4]    = {515,256,512,1024};
  int sa1_c[3][3], sa1_p[3][3], sa2_c[2][3], sa2_p[2][3], sa3_c[3], sa3_p[3];
  for (int m=0;m<3;m++) for (int l=0;l<3;l++) sa1_c[m][l]=addBN(SA1D[m][l],SA1D[m][l+1]);
  for (int m=0;m<3;m++) for (int l=0;l<3;l++) sa1_p[m][l]=addBN(SA1D[m][l],SA1D[m][l+1]);
  for (int m=0;m<2;m++) for (int l=0;l<3;l++) sa2_c[m][l]=addBN(SA2D[m][l],SA2D[m][l+1]);
  for (int m=0;m<2;m++) for (int l=0;l<3;l++) sa2_p[m][l]=addBN(SA2D[m][l],SA2D[m][l+1]);
  for (int l=0;l<3;l++) sa3_c[l]=addBN(SA3D[l],SA3D[l+1]);
  for (int l=0;l<3;l++) sa3_p[l]=addBN(SA3D[l],SA3D[l+1]);

  // ---- workspace bump allocator
  char* base = (char*)d_ws; size_t cur = 0;
  auto alo = [&](size_t bytes)->char*{
    char* p = base + cur;
    cur += (bytes + 255) & ~(size_t)255;
    return p;
  };
  float* l0x   = (float*)alo(sizeof(float)*(size_t)BATCH*NPTS*3);
  float* l0c   = (float*)alo(sizeof(float)*(size_t)BATCH*NPTS*3);
  half_t* wA   = (half_t*)alo(sizeof(half_t)*wCur);
  float* stA   = (float*)alo(sizeof(float)*sCur);
  int*   fps1  = (int*)  alo(sizeof(int)*(size_t)BATCH*512);
  float* l1xyz = (float*)alo(sizeof(float)*(size_t)BATCH*512*3);
  int*   idx1a = (int*)  alo(sizeof(int)*(size_t)BATCH*512*32);
  int*   idx1b = (int*)  alo(sizeof(int)*(size_t)BATCH*512*64);
  int*   idx1c = (int*)  alo(sizeof(int)*(size_t)BATCH*512*128);
  float* l1p   = (float*)alo(sizeof(float)*(size_t)BATCH*512*320);
  float* l1c   = (float*)alo(sizeof(float)*(size_t)BATCH*512*320);
  int*   fps2  = (int*)  alo(sizeof(int)*(size_t)BATCH*128);
  float* l2xyz = (float*)alo(sizeof(float)*(size_t)BATCH*128*3);
  int*   idx2a = (int*)  alo(sizeof(int)*(size_t)BATCH*128*64);
  int*   idx2b = (int*)  alo(sizeof(int)*(size_t)BATCH*128*128);
  float* l2p   = (float*)alo(sizeof(float)*(size_t)BATCH*128*512);
  float* l2c   = (float*)alo(sizeof(float)*(size_t)BATCH*128*512);
  float* g1    = (float*)alo(sizeof(float)*1024*256);
  float* g2    = (float*)alo(sizeof(float)*1024*512);
  float* g3    = (float*)alo(sizeof(float)*1024*1024);
  float* l3p   = (float*)alo(sizeof(float)*(size_t)BATCH*1024);
  float* l3c   = (float*)alo(sizeof(float)*(size_t)BATCH*1024);
  float* f3mid = (float*)alo(sizeof(float)*1024*256);
  float* l2pN  = (float*)alo(sizeof(float)*1024*256);
  float* l2cN  = (float*)alo(sizeof(float)*1024*256);
  float* ip2   = (float*)alo(sizeof(float)*(size_t)BATCH*512*256);
  float* f2mid = (float*)alo(sizeof(float)*4096*256);
  float* l1pN  = (float*)alo(sizeof(float)*4096*128);
  float* l1cN  = (float*)alo(sizeof(float)*4096*128);
  float* ip1   = (float*)alo(sizeof(float)*(size_t)BATCH*NPTS*128);
  float* f1mid = (float*)alo(sizeof(float)*(size_t)BATCH*NPTS*128);
  float* l0p   = (float*)alo(sizeof(float)*(size_t)BATCH*NPTS*128);
  float* l0cf  = (float*)alo(sizeof(float)*(size_t)BATCH*NPTS*128);
  float* feat  = (float*)alo(sizeof(float)*(size_t)BATCH*NPTS*128);
  float* logit = (float*)alo(sizeof(float)*(size_t)BATCH*NPTS*NCLS);

  // ---- weight/BN prep (deterministic each launch)
  for (int i=0;i<nLt;i++){
    LayerH& x = L[i];
    int tot = x.padOut * x.kpad;
    k_wconv<<<CDIV(tot,256),256,0,stream>>>(x.W, wA + x.wOff, x.Cin, x.Cout, x.kpad, x.padOut);
    k_stconv<<<CDIV(x.padOut,128),128,0,stream>>>(x.b, x.g, x.beta,
        stA + x.sOff, stA + x.sOff + x.padOut, x.Cout, x.padOut, x.mode);
  }

  auto mkFL = [&](int i)->FLayer{
    LayerH& x = L[i];
    FLayer f; f.Wt = wA + x.wOff; f.s = stA + x.sOff; f.t = stA + x.sOff + x.padOut;
    f.Cin=x.Cin; f.Cout=x.Cout; f.kpad=x.kpad; f.outW=x.padOut; return f;
  };
  auto saFused = [&](const int* li, const float* xyz, const float* src, int Csrc,
                     const int* idx, const float* ctr, float* out, int ostride, int ooff,
                     int N, int S, int K){
    FArgs A{}; A.nL = 3;
    int Wmax = 0;
    for (int i=0;i<3;i++){
      A.L[i]=mkFL(li[i]);
      if (A.L[i].kpad > Wmax) Wmax = A.L[i].kpad;
      if (A.L[i].outW > Wmax) Wmax = A.L[i].outW;
    }
    size_t lds = (size_t)2 * K * Wmax * sizeof(half_t);
    dim3 g(S, BATCH);
    k_sa_fused<<<g,128,lds,stream>>>(xyz,src,Csrc,idx,ctr,out,ostride,ooff,N,S,K,Wmax,A);
  };
  auto gemm = [&](int li, const float* Xa, int Ca, const float* Xb, int Cb, int xbDiv,
                  float* Y, int M, int relu){
    LayerH& x = L[li];
    dim3 g(CDIV(M,128), CDIV(x.Cout,128));
    k_gemm<<<g,256,0,stream>>>(Xa,Ca,Xb,Cb,xbDiv, wA + x.wOff, x.kpad,
        stA + x.sOff, stA + x.sOff + x.padOut, Y, M, x.Cout, relu);
  };

  // ---- stage 0: split input
  k_split<<<CDIV(BATCH*NPTS,256),256,0,stream>>>((const float*)d_in[0], l0x, l0c, NPTS);

  // ---- SA1 (npoint=512, radii .1/.2/.4, K=32/64/128)
  k_fps<<<BATCH,256,0,stream>>>(l0x, fps1, NPTS, 512);
  k_gather3<<<CDIV(BATCH*512,256),256,0,stream>>>(l0x, fps1, l1xyz, NPTS, 512);
  k_qball<<<CDIV(BATCH*512,128),128,0,stream>>>(l0x, l1xyz, idx1a, NPTS, 512, 32,  0.01f);
  k_qball<<<CDIV(BATCH*512,128),128,0,stream>>>(l0x, l1xyz, idx1b, NPTS, 512, 64,  0.04f);
  k_qball<<<CDIV(BATCH*512,128),128,0,stream>>>(l0x, l1xyz, idx1c, NPTS, 512, 128, 0.16f);
  saFused(sa1_p[0], l0x, l0x, 3, idx1a, l1xyz, l1p, 320, 0,   NPTS, 512, 32);
  saFused(sa1_p[1], l0x, l0x, 3, idx1b, l1xyz, l1p, 320, 64,  NPTS, 512, 64);
  saFused(sa1_p[2], l0x, l0x, 3, idx1c, l1xyz, l1p, 320, 192, NPTS, 512, 128);
  saFused(sa1_c[0], l0x, l0c, 3, idx1a, l1xyz, l1c, 320, 0,   NPTS, 512, 32);
  saFused(sa1_c[1], l0x, l0c, 3, idx1b, l1xyz, l1c, 320, 64,  NPTS, 512, 64);
  saFused(sa1_c[2], l0x, l0c, 3, idx1c, l1xyz, l1c, 320, 192, NPTS, 512, 128);

  // ---- SA2 (npoint=128, radii .4/.8, K=64/128)
  k_fps<<<BATCH,256,0,stream>>>(l1xyz, fps2, 512, 128);
  k_gather3<<<CDIV(BATCH*128,256),256,0,stream>>>(l1xyz, fps2, l2xyz, 512, 128);
  k_qball<<<CDIV(BATCH*128,128),128,0,stream>>>(l1xyz, l2xyz, idx2a, 512, 128, 64,  0.16f);
  k_qball<<<CDIV(BATCH*128,128),128,0,stream>>>(l1xyz, l2xyz, idx2b, 512, 128, 128, 0.64f);
  saFused(sa2_p[0], l1xyz, l1p, 320, idx2a, l2xyz, l2p, 512, 0,   512, 128, 64);
  saFused(sa2_p[1], l1xyz, l1p, 320, idx2b, l2xyz, l2p, 512, 256, 512, 128, 128);
  saFused(sa2_c[0], l1xyz, l1c, 320, idx2a, l2xyz, l2c, 512, 0,   512, 128, 64);
  saFused(sa2_c[1], l1xyz, l1c, 320, idx2b, l2xyz, l2c, 512, 256, 512, 128, 128);

  // ---- SA3 group-all (flat GEMM chain, M = B*128 = 1024)
  gemm(sa3_p[0], l2xyz, 3, l2p, 512, 0, g1, 1024, 1);
  gemm(sa3_p[1], g1, 256, nullptr, 0, 0, g2, 1024, 1);
  gemm(sa3_p[2], g2, 512, nullptr, 0, 0, g3, 1024, 1);
  k_maxpool<<<CDIV(BATCH*1024,256),256,0,stream>>>(g3, l3p, 128, 1024);
  gemm(sa3_c[0], l2xyz, 3, l2c, 512, 0, g1, 1024, 1);
  gemm(sa3_c[1], g1, 256, nullptr, 0, 0, g2, 1024, 1);
  gemm(sa3_c[2], g2, 512, nullptr, 0, 0, g3, 1024, 1);
  k_maxpool<<<CDIV(BATCH*1024,256),256,0,stream>>>(g3, l3c, 128, 1024);

  // ---- FP3 (S==1 broadcast via xbDiv=128)
  gemm(fp3_p[0], l2p, 512, l3p, 1024, 128, f3mid, 1024, 1);
  gemm(fp3_p[1], f3mid, 256, nullptr, 0, 0, l2pN, 1024, 1);
  gemm(fp3_c[0], l2c, 512, l3c, 1024, 128, f3mid, 1024, 1);
  gemm(fp3_c[1], f3mid, 256, nullptr, 0, 0, l2cN, 1024, 1);

  // ---- FP2 (3-NN interp 512<-128)
  k_interp3<<<CDIV(BATCH*512,128),128,0,stream>>>(l1xyz, l2xyz, l2pN, ip2, 512, 128, 256);
  gemm(fp2_p[0], l1p, 320, ip2, 256, 0, f2mid, 4096, 1);
  gemm(fp2_p[1], f2mid, 256, nullptr, 0, 0, l1pN, 4096, 1);
  k_interp3<<<CDIV(BATCH*512,128),128,0,stream>>>(l1xyz, l2xyz, l2cN, ip2, 512, 128, 256);
  gemm(fp2_c[0], l1c, 320, ip2, 256, 0, f2mid, 4096, 1);
  gemm(fp2_c[1], f2mid, 256, nullptr, 0, 0, l1cN, 4096, 1);

  // ---- FP1 (3-NN interp 8192<-512)
  k_interp3<<<CDIV(BATCH*NPTS,128),128,0,stream>>>(l0x, l1xyz, l1pN, ip1, NPTS, 512, 128);
  gemm(fp1_p[0], l0x, 3, ip1, 128, 0, f1mid, BATCH*NPTS, 1);
  gemm(fp1_p[1], f1mid, 128, nullptr, 0, 0, l0p, BATCH*NPTS, 1);
  k_interp3<<<CDIV(BATCH*NPTS,128),128,0,stream>>>(l0x, l1xyz, l1cN, ip1, NPTS, 512, 128);
  gemm(fp1_c[0], l0c, 3, ip1, 128, 0, f1mid, BATCH*NPTS, 1);
  gemm(fp1_c[1], f1mid, 128, nullptr, 0, 0, l0cf, BATCH*NPTS, 1);

  // ---- head + classifier + log_softmax
  gemm(head0,  l0p, 128, l0cf, 128, 0, feat, BATCH*NPTS, 1);
  gemm(conv2L, feat, 128, nullptr, 0, 0, logit, BATCH*NPTS, 0);
  k_logsoftmax<<<CDIV(BATCH*NPTS,128),128,0,stream>>>(logit, (float*)d_out, BATCH*NPTS, NCLS);

  // ---- l3 output (B,2048,1) appended after x
  k_l3copy<<<CDIV(BATCH*2048,256),256,0,stream>>>(l3p, l3c,
      (float*)d_out + (size_t)BATCH*NPTS*NCLS);
}
